// ClusteringMMD_63204738728430
// MI455X (gfx1250) — 8-bit matrix operand order repaired and checked by reading, compiled, not run on hardware
//
#include <hip/hip_runtime.h>
#include <stdint.h>

typedef __attribute__((ext_vector_type(16))) int   v16i;
typedef __attribute__((ext_vector_type(8)))  float v8f;

#define BATCH 128
#define NN 512
#define BINS 100

__global__ __launch_bounds__(256)
void convert_kernel(const float* __restrict__ adj,
                    unsigned char* __restrict__ a8,
                    float* __restrict__ deg) {
    const int waveId = (blockIdx.x * blockDim.x + threadIdx.x) >> 5;
    const int lane   = threadIdx.x & 31;
    const float4* src = (const float4*)(adj + (size_t)waveId * NN) + lane * 4;

    unsigned int packed[4];
    int cnt = 0;
#pragma unroll
    for (int q = 0; q < 4; ++q) {
        float4 f = src[q];
        unsigned int p = 0;
        if (f.x != 0.0f) { p |= 0x00000038u; ++cnt; }
        if (f.y != 0.0f) { p |= 0x00003800u; ++cnt; }
        if (f.z != 0.0f) { p |= 0x00380000u; ++cnt; }
        if (f.w != 0.0f) { p |= 0x38000000u; ++cnt; }
        packed[q] = p;
    }
    uint4 st; st.x = packed[0]; st.y = packed[1]; st.z = packed[2]; st.w = packed[3];
    ((uint4*)(a8 + (size_t)waveId * NN))[lane] = st;

    float fc = (float)cnt;
    fc += __shfl_xor(fc, 1, 32);
    fc += __shfl_xor(fc, 2, 32);
    fc += __shfl_xor(fc, 4, 32);
    fc += __shfl_xor(fc, 8, 32);
    fc += __shfl_xor(fc, 16, 32);
    if (lane == 0) deg[waveId] = fc;
}

__global__ __launch_bounds__(256)
void tri2_kernel(const unsigned char* __restrict__ a8,
                 float* __restrict__ tri2) {
    const int b    = blockIdx.y;
    const int wave = threadIdx.x >> 5;
    const int lane = threadIdx.x & 31;
    const int tile = blockIdx.x * 8 + wave;
    const int tr   = tile >> 5;
    const int tc   = tile & 31;
    const size_t base = (size_t)b * (NN * NN);

    const int m  = lane & 15;
    const int hf = lane >> 4;
    const int rowA = tr * 16 + m;
    const int rowB = tc * 16 + m;
    const uint32_t* a8w = (const uint32_t*)(a8 + base);

    v8f c = {};
#pragma unroll
    for (int ks = 0; ks < 4; ++ks) {
        const int K0 = ks * 128;
        v16i A, Bm;
#pragma unroll
        for (int v = 0; v < 16; ++v) {
            int kk = ((v & 8) << 3) + (((v & 7) >> 1) << 4) + ((v & 1) << 2)
                   + (hf << 3) + K0;
            A[v] = (int)a8w[((size_t)rowA * NN + kk) >> 2];
        }
#pragma unroll
        for (int v = 0; v < 16; ++v) {
            int kk = ((v & 8) << 3) + (((v & 7) >> 1) << 4) + ((v & 1) << 2)
                   + (hf << 3) + K0;
            Bm[v] = (int)a8w[((size_t)rowB * NN + kk) >> 2];
        }
        c = __builtin_amdgcn_wmma_f32_16x16x128_fp8_fp8(A, Bm, (short)0, c,
                                                        false, false);
    }

    const int col = tc * 16 + (lane & 15);
#pragma unroll
    for (int r = 0; r < 8; ++r) {
        const int row = tr * 16 + hf * 8 + r;
        float v = (a8[base + (size_t)row * NN + col] != 0) ? c[r] : 0.0f;
        v += __shfl_xor(v, 1, 32);
        v += __shfl_xor(v, 2, 32);
        v += __shfl_xor(v, 4, 32);
        v += __shfl_xor(v, 8, 32);
        if ((lane & 15) == 0)
            atomicAdd(&tri2[b * NN + row], v);
    }
}

__global__ __launch_bounds__(128)
void hist_kernel(const float* __restrict__ tri2,
                 const float* __restrict__ deg,
                 float* __restrict__ hist) {
    __shared__ float bins[BINS];
    const int b = blockIdx.x;
    for (int i = threadIdx.x; i < BINS; i += blockDim.x) bins[i] = 0.0f;
    __syncthreads();
    for (int i = threadIdx.x; i < NN; i += blockDim.x) {
        float d = deg[b * NN + i];
        float denom = d * (d - 1.0f);
        float t = tri2[b * NN + i];
        float cc = (denom > 0.0f) ? (t / fmaxf(denom, 1.0f)) : 0.0f;
        int idx = (int)(cc * (float)BINS);
        idx = min(max(idx, 0), BINS - 1);
        atomicAdd(&bins[idx], 1.0f);
    }
    __syncthreads();
    for (int i = threadIdx.x; i < BINS; i += blockDim.x)
        hist[b * BINS + i] = bins[i];
}

__global__ __launch_bounds__(256)
void mmd_kernel(const float* __restrict__ h1,
                const float* __restrict__ h2,
                float* __restrict__ acc) {
    const int combo = blockIdx.x;
    const float* X = (combo == 1) ? h2 : h1;
    const float* Y = (combo == 0) ? h1 : h2;
    float s = 0.0f;
    for (int p = threadIdx.x; p < BATCH * BATCH; p += blockDim.x) {
        const float* xa = X + (p >> 7) * BINS;
        const float* yb = Y + (p & 127) * BINS;
        float sq = 0.0f;
        for (int k = 0; k < BINS; ++k) {
            float d = xa[k] - yb[k];
            sq += d * d;
        }
        s += __expf(-0.5f * sq);
    }
    __shared__ float red[256];
    red[threadIdx.x] = s;
    __syncthreads();
    for (int st = 128; st > 0; st >>= 1) {
        if ((int)threadIdx.x < st) red[threadIdx.x] += red[threadIdx.x + st];
        __syncthreads();
    }
    if (threadIdx.x == 0) atomicAdd(&acc[combo], red[0]);
}

__global__ void final_kernel(const float* __restrict__ acc,
                             float* __restrict__ out) {
    const float inv = 1.0f / (float)(BATCH * BATCH);
    out[0] = acc[0] * inv + acc[1] * inv - 2.0f * acc[2] * inv;
}

extern "C" void kernel_launch(void* const* d_in, const int* in_sizes, int n_in,
                              void* d_out, int out_size, void* d_ws, size_t ws_size,
                              hipStream_t stream) {
    const float* adj1 = (const float*)d_in[0];
    const float* adj2 = (const float*)d_in[1];

    uint8_t* ws = (uint8_t*)d_ws;
    const size_t A8_BYTES = (size_t)BATCH * NN * NN;
    unsigned char* a8_1 = ws;
    unsigned char* a8_2 = ws + A8_BYTES;
    float* tri2_1 = (float*)(ws + 2 * A8_BYTES);
    float* tri2_2 = tri2_1 + BATCH * NN;
    float* deg_1  = tri2_2 + BATCH * NN;
    float* deg_2  = deg_1  + BATCH * NN;
    float* hist1  = deg_2  + BATCH * NN;
    float* hist2  = hist1  + BATCH * BINS;
    float* acc    = hist2  + BATCH * BINS;

    hipMemsetAsync(tri2_1, 0, 2 * (size_t)BATCH * NN * sizeof(float), stream);
    hipMemsetAsync(acc, 0, 3 * sizeof(float), stream);

    const int convBlocks = (BATCH * NN) / 8;
    convert_kernel<<<convBlocks, 256, 0, stream>>>(adj1, a8_1, deg_1);
    convert_kernel<<<convBlocks, 256, 0, stream>>>(adj2, a8_2, deg_2);

    dim3 g(128, BATCH);
    tri2_kernel<<<g, 256, 0, stream>>>(a8_1, tri2_1);
    tri2_kernel<<<g, 256, 0, stream>>>(a8_2, tri2_2);

    hist_kernel<<<BATCH, 128, 0, stream>>>(tri2_1, deg_1, hist1);
    hist_kernel<<<BATCH, 128, 0, stream>>>(tri2_2, deg_2, hist2);

    mmd_kernel<<<3, 256, 0, stream>>>(hist1, hist2, acc);
    final_kernel<<<1, 1, 0, stream>>>(acc, (float*)d_out);
}
